// BaseAttentionBlock_13804024889551
// MI455X (gfx1250) — matrix operand order repaired and checked by reading, not compiled, not run on hardware
//
#include <hip/hip_runtime.h>

// ---------------------------------------------------------------------------
// Types for CDNA5 WMMA (gfx1250, wave32)
// ---------------------------------------------------------------------------
typedef __attribute__((ext_vector_type(16))) __bf16 bf16x16;
typedef __attribute__((ext_vector_type(8)))  float  float8;

union AFrag {
    bf16x16 v;
    uint4   q[2];
};

static __device__ __forceinline__ float8 wmma_bf16(bf16x16 a, bf16x16 b, float8 c) {
    // (neg_a, A, neg_b, B, c_mod, C, reuse_a, reuse_b)
    return __builtin_amdgcn_wmma_f32_16x16x32_bf16(false, a, false, b, (short)0, c,
                                                   false, false);
}

// Problem constants (from reference): B=8, C=512, H=W=64 -> N=4096,
// K=V=256, O=512.
#define NB   8
#define CIN  512
#define NPOS 4096
#define KC   256
#define OC   512
#define BN_EPS 1e-5f

// ---------------------------------------------------------------------------
// Kernel 0: prep. Fold BN into wk, pack [wk'; wv] -> bf16 [512][512],
// bf16-ize wW, fuse biases.
// ---------------------------------------------------------------------------
__global__ __launch_bounds__(256) void prep_kernel(
    const float* __restrict__ wv, const float* __restrict__ bv,
    const float* __restrict__ wk, const float* __restrict__ bk,
    const float* __restrict__ gamma, const float* __restrict__ beta,
    const float* __restrict__ rmean, const float* __restrict__ rvar,
    const float* __restrict__ wW,
    __bf16* __restrict__ wcomb, __bf16* __restrict__ wWbf,
    float* __restrict__ bcomb)
{
    int idx = blockIdx.x * 256 + threadIdx.x;
    if (idx < 512 * 512) {
        int row = idx >> 9, c = idx & 511;
        float val;
        if (row < 256) {
            float s = gamma[row] * rsqrtf(rvar[row] + BN_EPS);
            val = wk[row * 512 + c] * s;
        } else {
            val = wv[(row - 256) * 512 + c];
        }
        wcomb[idx] = (__bf16)val;
    }
    if (idx < 512 * 256) wWbf[idx] = (__bf16)wW[idx];
    if (idx < 512) {
        float bval;
        if (idx < 256) {
            float s = gamma[idx] * rsqrtf(rvar[idx] + BN_EPS);
            bval = (bk[idx] - rmean[idx]) * s + beta[idx];
        } else {
            bval = bv[idx - 256];
        }
        bcomb[idx] = bval;
    }
}

// ---------------------------------------------------------------------------
// Kernel 1: fused key/value projection GEMM.
//   rows 0..255  : key   -> BN-folded weights, +bias, ReLU, store kbuf[B][N][256] (transposed)
//   rows 256..511: value -> +bias,           store vbuf[B][256][N]
// Tiling: WG = 8 waves, M-block 128 channels, N-block 64 positions, K over C=512 by 32.
// ---------------------------------------------------------------------------
__global__ __launch_bounds__(256) void proj_kernel(
    const float* __restrict__ x,
    const __bf16* __restrict__ wcomb, const float* __restrict__ bcomb,
    __bf16* __restrict__ kbuf, __bf16* __restrict__ vbuf)
{
    __shared__ __bf16 At[128 * 32];   // [m][c] weight tile
    __shared__ __bf16 Bt[64 * 32];    // [n][c] x tile (transposed, bf16)

    int bx = blockIdx.x;
    int mb = bx & 3;
    int ntile = (bx >> 2) & 63;
    int b = bx >> 8;
    int nbase = ntile * 64;

    int t = threadIdx.x;
    int lane = t & 31, w = t >> 5;
    int half = lane >> 4, l15 = lane & 15;

    const float* xb = x + (size_t)b * CIN * NPOS;
    const uint4* wc4 = (const uint4*)wcomb;
    uint4* At4 = (uint4*)At;
    uint4* Bt4 = (uint4*)Bt;

    float8 acc[4];
#pragma unroll
    for (int j = 0; j < 4; ++j) acc[j] = (float8)0.0f;

    for (int c0 = 0; c0 < CIN; c0 += 32) {
        __syncthreads();
        {   // A tile: 128 rows x 4 uint4/row = 512 uint4, 2 per thread
            int row = t >> 1;
            int part = (t & 1) * 2;
#pragma unroll
            for (int j = 0; j < 2; ++j)
                At4[row * 4 + part + j] =
                    wc4[(size_t)(mb * 128 + row) * 64 + (c0 >> 3) + part + j];
        }
        {   // B tile: [64 n][32 c], transpose+convert from x[c][n]
            int n = t & 63;
            int cg = (t >> 6) * 8;
#pragma unroll
            for (int j = 0; j < 8; ++j) {
                int c = cg + j;
                Bt[n * 32 + c] = (__bf16)xb[(size_t)(c0 + c) * NPOS + nbase + n];
            }
        }
        __syncthreads();

        AFrag a;
        a.q[0] = At4[(w * 16 + l15) * 4 + half];
        a.q[1] = At4[(w * 16 + l15) * 4 + 2 + half];
#pragma unroll
        for (int j = 0; j < 4; ++j) {
            AFrag bb;
            int r = j * 16 + l15;
            bb.q[0] = Bt4[r * 4 + half];
            bb.q[1] = Bt4[r * 4 + 2 + half];
            acc[j] = wmma_bf16(a.v, bb.v, acc[j]);
        }
    }

    // Epilogue: C-layout element (M = r + 8*half, Ncol = l15)
    int mrow0 = mb * 128 + w * 16;
#pragma unroll
    for (int j = 0; j < 4; ++j) {
        int n = nbase + j * 16 + l15;
#pragma unroll
        for (int r = 0; r < 8; ++r) {
            int ch = mrow0 + r + 8 * half;
            float val = acc[j][r] + bcomb[ch];
            if (ch < 256) {
                val = fmaxf(val, 0.0f);  // ReLU (key path)
                kbuf[((size_t)b * NPOS + n) * KC + ch] = (__bf16)val;
            } else {
                vbuf[((size_t)b * KC + (ch - 256)) * NPOS + n] = (__bf16)val;
            }
        }
    }
}

// ---------------------------------------------------------------------------
// Kernel 2: flash attention. sim = (k^T k) * 1/16, softmax(m), ctx = P V^T.
// WG = 8 waves; each wave owns 16 query rows and all 256 v-channels.
// Dynamic LDS: Qs 64KB | Ks 16KB | Vs 16KB | Ps 8KB  (104KB)
// ---------------------------------------------------------------------------
__global__ __launch_bounds__(256) void attn_kernel(
    const __bf16* __restrict__ kbuf, const __bf16* __restrict__ vbuf,
    __bf16* __restrict__ ctx)
{
    extern __shared__ char smem[];
    __bf16* Qs = (__bf16*)(smem);             // [128][256]
    __bf16* Ks = (__bf16*)(smem + 65536);     // [32][256]   (m-major)
    __bf16* Vs = (__bf16*)(smem + 81920);     // [256][32]   (v-major)
    __bf16* Ps = (__bf16*)(smem + 98304);     // per-wave [16][32]

    int bx = blockIdx.x;
    int b = bx >> 5;
    int nb = bx & 31;

    int t = threadIdx.x;
    int lane = t & 31, w = t >> 5;
    int half = lane >> 4, l15 = lane & 15;

    const uint4* kb4 = (const uint4*)(kbuf + (size_t)b * NPOS * KC);  // [N][32 uint4]
    const uint4* vb4 = (const uint4*)(vbuf + (size_t)b * KC * NPOS);  // [V][512 uint4]
    uint4* Qs4 = (uint4*)Qs;
    uint4* Ks4 = (uint4*)Ks;
    uint4* Vs4 = (uint4*)Vs;
    uint4* Ps4 = (uint4*)(Ps + w * (16 * 32));

    {   // load Q strip: 128 rows x 32 uint4 -> 16 uint4 per thread
        int row = t >> 1;
        int part = (t & 1) * 16;
#pragma unroll
        for (int j = 0; j < 16; ++j)
            Qs4[row * 32 + part + j] = kb4[(size_t)(nb * 128 + row) * 32 + part + j];
    }

    float8 accv[16];
#pragma unroll
    for (int v = 0; v < 16; ++v) accv[v] = (float8)0.0f;
    float mrow[8], lrow[8];
#pragma unroll
    for (int r = 0; r < 8; ++r) { mrow[r] = -3.0e38f; lrow[r] = 0.0f; }

    for (int m0 = 0; m0 < NPOS; m0 += 32) {
        __syncthreads();
        {   // K tile: 32 rows x 32 uint4 -> 4 uint4/thread
            int row = t >> 3;
            int part = (t & 7) * 4;
#pragma unroll
            for (int j = 0; j < 4; ++j)
                Ks4[row * 32 + part + j] = kb4[(size_t)(m0 + row) * 32 + part + j];
        }
        {   // V tile: 256 rows x 4 uint4 -> one v-row per thread
#pragma unroll
            for (int j = 0; j < 4; ++j)
                Vs4[t * 4 + j] = vb4[(size_t)t * 512 + (m0 >> 3) + j];
        }
        __syncthreads();

        if (m0 + 32 < NPOS)  // warm next k-tile in L2 (global_prefetch_b8)
            __builtin_prefetch(&kb4[(size_t)(m0 + 32 + (t >> 3)) * 32], 0, 1);

        // ---- S = Q^T K : two 16x16 f32 tiles over kc = 8 chunks of 32 ----
        float8 s0 = (float8)0.0f, s1 = (float8)0.0f;
#pragma unroll
        for (int c = 0; c < 8; ++c) {
            AFrag a;
            int qr = (w * 16 + l15) * 32 + c * 4;
            a.q[0] = Qs4[qr + half];
            a.q[1] = Qs4[qr + 2 + half];
            AFrag b0, b1;
            b0.q[0] = Ks4[l15 * 32 + c * 4 + half];
            b0.q[1] = Ks4[l15 * 32 + c * 4 + 2 + half];
            b1.q[0] = Ks4[(16 + l15) * 32 + c * 4 + half];
            b1.q[1] = Ks4[(16 + l15) * 32 + c * 4 + 2 + half];
            s0 = wmma_bf16(a.v, b0.v, s0);
            s1 = wmma_bf16(a.v, b1.v, s1);
        }

        // ---- online softmax over this 16x32 S slab ----
#pragma unroll
        for (int r = 0; r < 8; ++r) {
            float e0 = s0[r] * 0.0625f;   // * K^-0.5 = 1/16
            float e1 = s1[r] * 0.0625f;
            float tm = fmaxf(e0, e1);
#pragma unroll
            for (int off = 1; off < 16; off <<= 1)
                tm = fmaxf(tm, __shfl_xor(tm, off, 16));
            float mnew = fmaxf(mrow[r], tm);
            float corr = __expf(mrow[r] - mnew);
            float p0 = __expf(e0 - mnew);
            float p1 = __expf(e1 - mnew);
            float rs = p0 + p1;
#pragma unroll
            for (int off = 1; off < 16; off <<= 1)
                rs += __shfl_xor(rs, off, 16);
            lrow[r] = lrow[r] * corr + rs;
            mrow[r] = mnew;
            int prow = r + 8 * half;
            Ps[w * 512 + prow * 32 + l15]      = (__bf16)p0;
            Ps[w * 512 + prow * 32 + 16 + l15] = (__bf16)p1;
#pragma unroll
            for (int v = 0; v < 16; ++v) accv[v][r] *= corr;
        }
        // Ps is per-wave private; ensure ds stores land before fragment reads.
        asm volatile("s_wait_dscnt 0" ::: "memory");

        // ---- ctx += P @ V^T over 16 v-tiles ----
        AFrag pa;
        pa.q[0] = Ps4[l15 * 4 + half];
        pa.q[1] = Ps4[l15 * 4 + 2 + half];
#pragma unroll
        for (int v = 0; v < 16; ++v) {
            AFrag vbf;
            vbf.q[0] = Vs4[(v * 16 + l15) * 4 + half];
            vbf.q[1] = Vs4[(v * 16 + l15) * 4 + 2 + half];
            accv[v] = wmma_bf16(pa.v, vbf.v, accv[v]);
        }
    }

    // ---- finalize: ctx[b][n][vc] = acc / l  (bf16, position-major) ----
#pragma unroll
    for (int r = 0; r < 8; ++r) {
        float inv = 1.0f / lrow[r];
        int n = nb * 128 + w * 16 + r + 8 * half;
#pragma unroll
        for (int v = 0; v < 16; ++v) {
            int vc = v * 16 + l15;
            ctx[((size_t)b * NPOS + n) * KC + vc] = (__bf16)(accv[v][r] * inv);
        }
    }
}

// ---------------------------------------------------------------------------
// Kernel 3: output 1x1 conv. out[b][o][n] = wW[o][:] . ctx[b][n][:] + bW[o]
// Same tiling as proj: M-block 128 (of 512), N-block 64, K over V=256 by 32.
// ---------------------------------------------------------------------------
__global__ __launch_bounds__(256) void out_kernel(
    const __bf16* __restrict__ ctx, const __bf16* __restrict__ wWbf,
    const float* __restrict__ bW, float* __restrict__ out)
{
    __shared__ __bf16 At[128 * 32];  // [o][v]
    __shared__ __bf16 Bt[64 * 32];   // [n][v]

    int bx = blockIdx.x;
    int mb = bx & 3;
    int ntile = (bx >> 2) & 63;
    int b = bx >> 8;
    int nbase = ntile * 64;

    int t = threadIdx.x;
    int lane = t & 31, w = t >> 5;
    int half = lane >> 4, l15 = lane & 15;

    const uint4* wW4 = (const uint4*)wWbf;                     // [512][32 uint4]
    const uint4* cx4 = (const uint4*)(ctx + (size_t)b * NPOS * KC);  // [N][32 uint4]
    uint4* At4 = (uint4*)At;
    uint4* Bt4 = (uint4*)Bt;

    float8 acc[4];
#pragma unroll
    for (int j = 0; j < 4; ++j) acc[j] = (float8)0.0f;

    for (int c0 = 0; c0 < KC; c0 += 32) {
        __syncthreads();
        {   // A tile
            int row = t >> 1;
            int part = (t & 1) * 2;
#pragma unroll
            for (int j = 0; j < 2; ++j)
                At4[row * 4 + part + j] =
                    wW4[(size_t)(mb * 128 + row) * 32 + (c0 >> 3) + part + j];
        }
        {   // B tile: 64 rows x 4 uint4
            int row = t >> 2;
            int part = t & 3;
            Bt4[row * 4 + part] =
                cx4[(size_t)(nbase + row) * 32 + (c0 >> 3) + part];
        }
        __syncthreads();

        AFrag a;
        a.q[0] = At4[(w * 16 + l15) * 4 + half];
        a.q[1] = At4[(w * 16 + l15) * 4 + 2 + half];
#pragma unroll
        for (int j = 0; j < 4; ++j) {
            AFrag bb;
            int r = j * 16 + l15;
            bb.q[0] = Bt4[r * 4 + half];
            bb.q[1] = Bt4[r * 4 + 2 + half];
            acc[j] = wmma_bf16(a.v, bb.v, acc[j]);
        }
    }

    int orow0 = mb * 128 + w * 16;
#pragma unroll
    for (int j = 0; j < 4; ++j) {
        int n = nbase + j * 16 + l15;
#pragma unroll
        for (int r = 0; r < 8; ++r) {
            int o = orow0 + r + 8 * half;
            out[((size_t)b * OC + o) * NPOS + n] = acc[j][r] + bW[o];
        }
    }
}

// ---------------------------------------------------------------------------
// Launch
// ---------------------------------------------------------------------------
extern "C" void kernel_launch(void* const* d_in, const int* in_sizes, int n_in,
                              void* d_out, int out_size, void* d_ws, size_t ws_size,
                              hipStream_t stream) {
    const float* x     = (const float*)d_in[0];
    const float* wv    = (const float*)d_in[1];
    const float* bv    = (const float*)d_in[2];
    const float* wk    = (const float*)d_in[3];
    const float* bk    = (const float*)d_in[4];
    const float* gamma = (const float*)d_in[5];
    const float* beta  = (const float*)d_in[6];
    const float* rmean = (const float*)d_in[7];
    const float* rvar  = (const float*)d_in[8];
    const float* wW    = (const float*)d_in[9];
    const float* bW    = (const float*)d_in[10];
    float* out = (float*)d_out;

    // Workspace layout (bytes)
    char* ws = (char*)d_ws;
    const size_t kv_bytes = (size_t)NB * NPOS * KC * 2;  // 16 MB each
    __bf16* kbuf  = (__bf16*)(ws);                       // [B][N][256] bf16
    __bf16* vbuf  = (__bf16*)(ws + kv_bytes);            // [B][256][N] bf16
    __bf16* ctx   = (__bf16*)(ws + 2 * kv_bytes);        // [B][N][256] bf16
    __bf16* wcomb = (__bf16*)(ws + 3 * kv_bytes);        // [512][512]  bf16
    __bf16* wWbf  = (__bf16*)(ws + 3 * kv_bytes + 512 * 512 * 2);
    float*  bcomb = (float*) (ws + 3 * kv_bytes + 512 * 512 * 2 + 512 * 256 * 2);

    prep_kernel<<<1024, 256, 0, stream>>>(wv, bv, wk, bk, gamma, beta, rmean, rvar,
                                          wW, wcomb, wWbf, bcomb);

    proj_kernel<<<NB * 64 * 4, 256, 0, stream>>>(x, wcomb, bcomb, kbuf, vbuf);

    (void)hipFuncSetAttribute((const void*)attn_kernel,
                              hipFuncAttributeMaxDynamicSharedMemorySize, 106496);
    attn_kernel<<<NB * 32, 256, 106496, stream>>>(kbuf, vbuf, ctx);

    out_kernel<<<NB * 64 * 4, 256, 0, stream>>>(ctx, wWbf, bW, out);
}
